// KGAT_2_raw_new_ver_70643622084957
// MI455X (gfx1250) — matrix operand order repaired and checked by reading, not compiled, not run on hardware
//
#include <hip/hip_runtime.h>
#include <hip/hip_bf16.h>

typedef __attribute__((ext_vector_type(16))) __bf16 v16bf;
typedef __attribute__((ext_vector_type(8)))  __bf16 v8bf;
typedef __attribute__((ext_vector_type(8)))  float  v8f;

union F16x16 { v16bf v; __bf16 e[16]; v8bf h[2]; };

__device__ __forceinline__ float lrelu(float x) { return x > 0.0f ? x : 0.01f * x; }

// Async global->LDS copy (gfx1250): guarded so absence degrades to sync path.
#if defined(__AMDGCN__) && __has_builtin(__builtin_amdgcn_global_load_async_to_lds_b128) && __has_builtin(__builtin_amdgcn_s_wait_asynccnt)
#define USE_ASYNC_LDS 1
#else
#define USE_ASYNC_LDS 0
#endif

#if USE_ASYNC_LDS
typedef int v4i __attribute__((vector_size(4 * sizeof(int))));
typedef __attribute__((address_space(1))) v4i v4i_g;   // global (prints __device__)
typedef __attribute__((address_space(3))) v4i v4i_l;   // LDS    (prints __shared__)
__device__ __forceinline__ void async_copy_b128(const void* g, const void* l)
{
    // generic->AS pointers via integer casts (low 32 bits of a generic LDS
    // address are the LDS byte offset per the CDNA5 aperture rules)
    __builtin_amdgcn_global_load_async_to_lds_b128(
        (v4i_g*)(unsigned long long)g,
        (v4i_l*)(unsigned int)(unsigned long long)l, 0, 0);
}
#endif

// ---------------------------------------------------------------------------
// side = A @ ego  (COO scatter; one wave per edge; b128 gathers, f32 atomics)
// ---------------------------------------------------------------------------
__global__ __launch_bounds__(256)
void spmm_scatter(const int* __restrict__ rows, const int* __restrict__ cols,
                  const float* __restrict__ vals, const float* __restrict__ x,
                  float* __restrict__ y, int E, int din)
{
    int edge = (blockIdx.x * blockDim.x + threadIdx.x) >> 5;
    int lane = threadIdx.x & 31;
    if (edge >= E) return;
    int r = rows[edge];
    int c = cols[edge];
    float v = vals[edge];
    const float4* xr = (const float4*)(x + (long)c * din);
    float*        yr = y + (long)r * din;
    int nvec = din >> 2;
    for (int d = lane; d < nvec; d += 32) {
        float4 xv = xr[d];
        float* yp = yr + d * 4;
        atomicAdd(yp + 0, v * xv.x);
        atomicAdd(yp + 1, v * xv.y);
        atomicAdd(yp + 2, v * xv.z);
        atomicAdd(yp + 3, v * xv.w);
    }
}

__global__ __launch_bounds__(256)
void zero_f32(float* __restrict__ p, long n)
{
    long stride = (long)gridDim.x * blockDim.x;
    for (long i = blockIdx.x * (long)blockDim.x + threadIdx.x; i < n; i += stride)
        p[i] = 0.0f;
}

__global__ __launch_bounds__(256)
void copy_embed(const float* __restrict__ ego, float* __restrict__ out, int N)
{
    long n = (long)N * 256;
    long stride = (long)gridDim.x * blockDim.x;
    for (long i = blockIdx.x * (long)blockDim.x + threadIdx.x; i < n; i += stride) {
        long row = i >> 8;
        int  col = (int)(i & 255);
        out[row * 704 + col] = ego[i];
    }
}

// ---------------------------------------------------------------------------
// Fused bi-interaction layer (bf16 WMMA, f32 accumulate):
//   ego_next = leaky((ego+side)@W1 + b1) + leaky((ego*side)@W2 + b2)
// Block = 4 waves. Wave tile = 16 rows x 32 cols (4 WMMAs per 32-K chunk).
// W panels async-staged to LDS, transposed to padded column-major bf16.
// ---------------------------------------------------------------------------
#define MAXD  256
#define BSTR  (MAXD + 8)      // bf16 column stride (pad avoids bank conflicts)
#define PASSR 128             // K-rows staged per async pass

__global__ __launch_bounds__(128)
void kgat_layer_wmma(const float* __restrict__ ego,
                     const float* __restrict__ side,
                     const float* __restrict__ W1, const float* __restrict__ b1,
                     const float* __restrict__ W2, const float* __restrict__ b2,
                     float* __restrict__ ego_next,
                     int N, int din, int dout)
{
    __shared__ __align__(16) __bf16 sB1[32 * BSTR];   // [col][K] bf16
    __shared__ __align__(16) __bf16 sB2[32 * BSTR];
#if USE_ASYNC_LDS
    __shared__ __align__(16) float  sScr[PASSR * 32]; // f32 staging scratch
#endif

    const int tid     = threadIdx.x;
    const int lane    = tid & 31;
    const int wave    = tid >> 5;
    const int colBase = blockIdx.y * 32;
    const int rowBase = blockIdx.x * 64 + wave * 16;

    // ---- Stage W1/W2 panels (cols colBase..+31, all K) into column-major bf16
    for (int m = 0; m < 2; ++m) {
        const float* Wm  = m ? W2 : W1;
        __bf16*      dst = m ? sB2 : sB1;
        for (int kb0 = 0; kb0 < din; kb0 += PASSR) {
            int R = din - kb0; if (R > PASSR) R = PASSR;
#if USE_ASYNC_LDS
            for (int q = tid; q < R * 8; q += 128) {          // b128 chunks
                int k = q >> 3, c4 = (q & 7) * 4;
                async_copy_b128(Wm + (long)(kb0 + k) * dout + colBase + c4,
                                &sScr[k * 32 + c4]);
            }
            __builtin_amdgcn_s_wait_asynccnt(0);
            __syncthreads();
            for (int idx = tid; idx < R * 32; idx += 128) {   // transpose+cvt
                int k = idx >> 5, c = idx & 31;
                dst[c * BSTR + kb0 + k] = (__bf16)sScr[k * 32 + c];
            }
            __syncthreads();                                  // scr reuse guard
#else
            for (int idx = tid; idx < R * 32; idx += 128) {
                int k = idx >> 5, c = idx & 31;
                dst[c * BSTR + kb0 + k] = (__bf16)Wm[(long)(kb0 + k) * dout + colBase + c];
            }
#endif
        }
    }
#if !USE_ASYNC_LDS
    __syncthreads();
#endif

    const int mloc = lane & 15;
    const int half = lane >> 4;             // 0: lanes 0-15, 1: lanes 16-31
    int row  = rowBase + mloc;
    int rowc = row < N ? row : N - 1;       // clamp OOB loads; EXEC stays full
    const float* egoR  = ego  + (long)rowc * din;
    const float* sideR = side + (long)rowc * din;

    v8f accS0 = {0.f,0.f,0.f,0.f,0.f,0.f,0.f,0.f};
    v8f accS1 = accS0, accB0 = accS0, accB1 = accS0;

    for (int k0 = 0; k0 < din; k0 += 32) {
        // --- A fragments: lane holds K = ka..ka+15 (ka=k0+16*half)
        const int ka = k0 + half * 16;
        const float4* e4 = (const float4*)(egoR  + ka);
        const float4* s4 = (const float4*)(sideR + ka);
        float eg[16], sd[16];
        *(float4*)&eg[0]  = e4[0]; *(float4*)&eg[4]  = e4[1];
        *(float4*)&eg[8]  = e4[2]; *(float4*)&eg[12] = e4[3];
        *(float4*)&sd[0]  = s4[0]; *(float4*)&sd[4]  = s4[1];
        *(float4*)&sd[8]  = s4[2]; *(float4*)&sd[12] = s4[3];
        F16x16 aS, aB;
        #pragma unroll
        for (int i = 0; i < 16; ++i) {
            aS.e[i] = (__bf16)(eg[i] + sd[i]);
            aB.e[i] = (__bf16)(eg[i] * sd[i]);
        }
        // --- B fragments: lane = column, K = kb..kb+15 (kb=k0+16*half); 2x b128
        const int kb = k0 + half * 16;
        F16x16 bW1a, bW1b, bW2a, bW2b;
        const __bf16* p1a = &sB1[mloc        * BSTR + kb];
        const __bf16* p1b = &sB1[(mloc + 16) * BSTR + kb];
        const __bf16* p2a = &sB2[mloc        * BSTR + kb];
        const __bf16* p2b = &sB2[(mloc + 16) * BSTR + kb];
        bW1a.h[0] = *(const v8bf*)p1a;       bW1a.h[1] = *(const v8bf*)(p1a + 8);
        bW1b.h[0] = *(const v8bf*)p1b;       bW1b.h[1] = *(const v8bf*)(p1b + 8);
        bW2a.h[0] = *(const v8bf*)p2a;       bW2a.h[1] = *(const v8bf*)(p2a + 8);
        bW2b.h[0] = *(const v8bf*)p2b;       bW2b.h[1] = *(const v8bf*)(p2b + 8);

        accS0 = __builtin_amdgcn_wmma_f32_16x16x32_bf16(false, aS.v, false, bW1a.v, (short)0, accS0, false, false);
        accS1 = __builtin_amdgcn_wmma_f32_16x16x32_bf16(false, aS.v, false, bW1b.v, (short)0, accS1, false, false);
        accB0 = __builtin_amdgcn_wmma_f32_16x16x32_bf16(false, aB.v, false, bW2a.v, (short)0, accB0, false, false);
        accB1 = __builtin_amdgcn_wmma_f32_16x16x32_bf16(false, aB.v, false, bW2b.v, (short)0, accB1, false, false);
    }

    // ---- Epilogue: bias + leaky_relu + sum; C/D layout VGPR r -> M=r+8*half
    const float bb1a = b1[colBase + mloc],      bb2a = b2[colBase + mloc];
    const float bb1b = b1[colBase + 16 + mloc], bb2b = b2[colBase + 16 + mloc];
    #pragma unroll
    for (int r = 0; r < 8; ++r) {
        int orow = rowBase + r + half * 8;
        if (orow < N) {
            float* op = ego_next + (long)orow * dout + colBase;
            op[mloc]      = lrelu(accS0[r] + bb1a) + lrelu(accB0[r] + bb2a);
            op[16 + mloc] = lrelu(accS1[r] + bb1b) + lrelu(accB1[r] + bb2b);
        }
    }
}

// ---------------------------------------------------------------------------
// Row L2-normalize into out[:, colOff:colOff+d]; one wave per row
// ---------------------------------------------------------------------------
__global__ __launch_bounds__(256)
void norm_store(const float* __restrict__ ego, float* __restrict__ out,
                int N, int d, int colOff)
{
    int row  = (blockIdx.x * blockDim.x + threadIdx.x) >> 5;
    int lane = threadIdx.x & 31;
    if (row >= N) return;
    const float* x = ego + (long)row * d;
    float ss = 0.0f;
    for (int j = lane; j < d; j += 32) { float v = x[j]; ss += v * v; }
    #pragma unroll
    for (int m = 16; m >= 1; m >>= 1) ss += __shfl_xor(ss, m, 32);
    float nrm = sqrtf(ss);
    nrm = nrm > 1e-12f ? nrm : 1e-12f;
    float inv = 1.0f / nrm;
    float* o = out + (long)row * 704 + colOff;
    for (int j = lane; j < d; j += 32) o[j] = x[j] * inv;
}

// ---------------------------------------------------------------------------
extern "C" void kernel_launch(void* const* d_in, const int* in_sizes, int n_in,
                              void* d_out, int out_size, void* d_ws, size_t ws_size,
                              hipStream_t stream)
{
    const float* ego0 = (const float*)d_in[0];
    const int*   Ar   = (const int*)  d_in[1];
    const int*   Ac   = (const int*)  d_in[2];
    const float* Av   = (const float*)d_in[3];
    const float* W1[3] = {(const float*)d_in[4],  (const float*)d_in[8],  (const float*)d_in[12]};
    const float* B1[3] = {(const float*)d_in[5],  (const float*)d_in[9],  (const float*)d_in[13]};
    const float* W2[3] = {(const float*)d_in[6],  (const float*)d_in[10], (const float*)d_in[14]};
    const float* B2[3] = {(const float*)d_in[7],  (const float*)d_in[11], (const float*)d_in[15]};

    const int N = in_sizes[0] / 256;
    const int E = in_sizes[1];

    float* out  = (float*)d_out;
    float* ws   = (float*)d_ws;
    float* side = ws;                               // N*256 f32
    float* bufA = ws + (size_t)N * 256;             // N*256 f32
    float* bufB = bufA + (size_t)N * 256;           // N*256 f32

    const int dims[4]   = {256, 256, 128, 64};
    const int colOff[3] = {256, 512, 640};

    copy_embed<<<2048, 256, 0, stream>>>(ego0, out, N);

    const float* cur = ego0;
    float*       nxt = bufA;
    for (int k = 0; k < 3; ++k) {
        const int din  = dims[k];
        const int dout = dims[k + 1];

        zero_f32<<<2048, 256, 0, stream>>>(side, (long)N * din);

        int sblocks = (int)(((long)E * 32 + 255) / 256);
        spmm_scatter<<<sblocks, 256, 0, stream>>>(Ar, Ac, Av, cur, side, E, din);

        dim3 grid((N + 63) / 64, dout / 32);
        kgat_layer_wmma<<<grid, 128, 0, stream>>>(cur, side,
                                                  W1[k], B1[k], W2[k], B2[k],
                                                  nxt, N, din, dout);

        int nblocks = (int)(((long)N * 32 + 255) / 256);
        norm_store<<<nblocks, 256, 0, stream>>>(nxt, out, N, dout, colOff[k]);

        cur = nxt;
        nxt = (nxt == bufA) ? bufB : bufA;
    }
}
